// SelfAttention_91242285236706
// MI455X (gfx1250) — matrix operand order repaired and checked by reading, not compiled, not run on hardware
//
#include <hip/hip_runtime.h>

// ---------------------------------------------------------------------------
// CDNA5 (gfx1250) wave32 WMMA attention block.
// All matmuls use v_wmma_f32_16x16x32_bf16 (fp32 accum, bf16 operands).
// ---------------------------------------------------------------------------

typedef __attribute__((ext_vector_type(16))) __bf16 v16bf;
typedef __attribute__((ext_vector_type(8)))  float  v8f;
typedef __attribute__((ext_vector_type(2)))  __bf16 v2bf;

__device__ __forceinline__ __bf16 f2bf(float f) { return (__bf16)f; }

__device__ __forceinline__ unsigned int pack2bf(float a, float b) {
  v2bf p;
  p.x = (__bf16)a;
  p.y = (__bf16)b;
  return __builtin_bit_cast(unsigned int, p);
}

// pack float4 -> 4 bf16 and store 8 bytes to LDS (dst must be 8B aligned)
__device__ __forceinline__ void store4bf(__bf16* dst, float4 v) {
  uint2 p;
  p.x = pack2bf(v.x, v.y);
  p.y = pack2bf(v.z, v.w);
  *reinterpret_cast<uint2*>(dst) = p;
}

// ISA 7.12.2: 16-bit A-matrix 16x32. lane: M = lane%16.
// VGPR0-3 hold K=0..7 (lanes 0-15) / K=8..15 (lanes 16-31); VGPR4-7 add +16.
__device__ __forceinline__ int kmapA(int i, int lane) {
  return ((i >> 3) << 4) + (((lane >> 4) & 1) << 3) + (i & 7);
}
// B-matrix 32x16: lane: N = lane%16; same K order as A: K=0..7 / 8..15 per half, elements 8-15 add +16.
__device__ __forceinline__ int kmapB(int i, int lane) {
  return ((i >> 3) << 4) + (((lane >> 4) & 1) << 3) + (i & 7);
}
// C/D 16x16 fp32: element f -> row M = f + 8*(lane>=16), col N = lane%16.

// ---------------------------------------------------------------------------
// GEMM: C[M,N] = A[M,K] * W[N,K]^T   (fp32 in/out, bf16 WMMA compute)
// 256 threads (8 waves), block tile 128x128, wave tile 32x64, K-step 32
// ---------------------------------------------------------------------------
#define GT_M 128
#define GT_N 128
#define GT_K 32
#define GT_PAD 4   // keep LDS rows 8B aligned for ds_store_b64

__global__ __launch_bounds__(256)
void gemm_bf16(const float* __restrict__ A, const float* __restrict__ W,
               float* __restrict__ C, int M, int N, int K) {
  __shared__ __bf16 As[GT_M][GT_K + GT_PAD];
  __shared__ __bf16 Ws[GT_N][GT_K + GT_PAD];

  const int tid  = threadIdx.x;
  const int lane = tid & 31;
  const int wave = tid >> 5;
  const int wm   = wave >> 1;   // 0..3 -> 32-row strip
  const int wn   = wave & 1;    // 0..1 -> 64-col strip
  const int mb   = blockIdx.y * GT_M;
  const int nb   = blockIdx.x * GT_N;
  const int ln   = lane & 15;
  const int hi   = (lane >> 4) & 1;

  const float4* A4 = reinterpret_cast<const float4*>(A);
  const float4* W4 = reinterpret_cast<const float4*>(W);
  const int K4 = K >> 2;

  v8f zero = {};
  v8f acc[2][4];
#pragma unroll
  for (int x = 0; x < 2; ++x)
#pragma unroll
    for (int y = 0; y < 4; ++y) acc[x][y] = zero;

  for (int k0 = 0; k0 < K; k0 += GT_K) {
    const int k04 = k0 >> 2;
    // stage A tile (128x32) fp32 -> bf16 : 1024 float4 over 256 threads
#pragma unroll
    for (int j = 0; j < 4; ++j) {
      int idx = j * 256 + tid;
      int r = idx >> 3, c4 = idx & 7;
      float4 v = A4[(size_t)(mb + r) * K4 + (k04 + c4)];
      store4bf(&As[r][c4 * 4], v);
    }
    // stage W tile (128x32)
#pragma unroll
    for (int j = 0; j < 4; ++j) {
      int idx = j * 256 + tid;
      int r = idx >> 3, c4 = idx & 7;
      float4 v = W4[(size_t)(nb + r) * K4 + (k04 + c4)];
      store4bf(&Ws[r][c4 * 4], v);
    }
    // prefetch next K tile (global_prefetch_b8)
    if (k0 + GT_K < K) {
      int r = tid >> 1, c = (tid & 1) << 4;
      __builtin_prefetch(&A[(size_t)(mb + r) * K + (k0 + GT_K + c)], 0, 1);
      __builtin_prefetch(&W[(size_t)(nb + r) * K + (k0 + GT_K + c)], 0, 1);
    }
    __syncthreads();

    v16bf af[2], bf[4];
#pragma unroll
    for (int t = 0; t < 2; ++t)
#pragma unroll
      for (int i = 0; i < 16; ++i)
        af[t][i] = As[wm * 32 + t * 16 + ln][kmapA(i, lane)];
#pragma unroll
    for (int y = 0; y < 4; ++y)
#pragma unroll
      for (int i = 0; i < 16; ++i)
        bf[y][i] = Ws[wn * 64 + y * 16 + ln][kmapB(i, lane)];

#pragma unroll
    for (int x = 0; x < 2; ++x)
#pragma unroll
      for (int y = 0; y < 4; ++y)
        acc[x][y] = __builtin_amdgcn_wmma_f32_16x16x32_bf16(
            false, af[x], false, bf[y], (short)0, acc[x][y], false, false);
    __syncthreads();
  }

#pragma unroll
  for (int x = 0; x < 2; ++x)
#pragma unroll
    for (int y = 0; y < 4; ++y)
#pragma unroll
      for (int f = 0; f < 8; ++f) {
        int m = mb + wm * 32 + x * 16 + f + 8 * hi;
        int n = nb + wn * 64 + y * 16 + ln;
        C[(size_t)m * N + n] = acc[x][y][f];
      }
}

// ---------------------------------------------------------------------------
// RoPE in place on (B,S,nheads,128) fp32; one thread per complex pair.
// ---------------------------------------------------------------------------
__global__ void rope_inplace(float* __restrict__ q, const float* __restrict__ fc,
                             const float* __restrict__ fs, const int* __restrict__ sp,
                             int total_pairs, int S, int nheads, int maxseq) {
  int sp0 = *sp;
  for (int idx = blockIdx.x * blockDim.x + threadIdx.x; idx < total_pairs;
       idx += gridDim.x * blockDim.x) {
    int i    = idx & 63;       // pair index (HD/2 = 64)
    int rest = idx >> 6;       // (b*S + s)*nheads + h
    int bs   = rest / nheads;
    int s    = bs % S;
    int p    = sp0 + s;
    float c = 1.f, sn = 0.f;
    if (p < maxseq) { c = fc[p * 64 + i]; sn = fs[p * 64 + i]; }
    size_t base = (size_t)rest * 128 + 2 * i;
    float xr = q[base], xi = q[base + 1];
    q[base]     = xr * c - xi * sn;
    q[base + 1] = xr * sn + xi * c;
  }
}

// ---------------------------------------------------------------------------
// Flash attention: block = 128 threads (4 waves) handles 64 query rows of one
// (b, head). K staged row-major, V staged TRANSPOSED in LDS so both score and
// PV fragments are contiguous LDS reads. GQA n_rep=4 shares chunks across waves.
// ---------------------------------------------------------------------------
__global__ __launch_bounds__(128)
void attn_kernel(const float* __restrict__ q,    // (B,S,32,128) roped
                 const float* __restrict__ kr,   // (B,S,8,128)  roped
                 const float* __restrict__ vr,   // (B,S,8,128)
                 const float* __restrict__ ck,   // (B,MS,8,128) cache (pre-roped)
                 const float* __restrict__ cv,   // (B,MS,8,128)
                 const int* __restrict__ sp,
                 float* __restrict__ out,        // (B,S,4096)
                 int B, int S, int MS) {
  constexpr int H = 32, KVH = 8, HD = 128;
  __shared__ __bf16 Ks[32][HD + 4];     // keys row-major (t, d)
  __shared__ __bf16 Vt[HD][36];         // values transposed (d, t)
  __shared__ __bf16 Pl[4][16][36];

  const int tid  = threadIdx.x;
  const int lane = tid & 31;
  const int wv_  = tid >> 5;               // wave 0..3
  const int nq = S >> 6;                   // 64-row query tiles
  const int qt = blockIdx.x % nq;
  const int h  = (blockIdx.x / nq) % H;
  const int b  = blockIdx.x / (nq * H);
  const int kvh = h >> 2;                  // n_rep = 4
  const int sp0 = *sp;
  const int hi  = (lane >> 4) & 1;
  const int ln  = lane & 15;
  const int qrow0 = qt * 64 + wv_ * 16;    // this wave's query rows
  const float scale = 0.08838834764831845f;    // 1/sqrt(128)

  // ---- Q fragments (16x128 -> 4 A-frags of 16x32), pre-scaled, b128 loads ----
  v16bf qa[4];
  {
    int srow = qrow0 + ln;
    const float4* qp4 = reinterpret_cast<const float4*>(
        q + (((size_t)(b * S + srow)) * H + h) * HD);
#pragma unroll
    for (int c = 0; c < 4; ++c) {
      int b0 = (c * 32 + 8 * hi) >> 2;
      float4 f0 = qp4[b0], f1 = qp4[b0 + 1];
      int b1 = (c * 32 + 16 + 8 * hi) >> 2;
      float4 f2 = qp4[b1], f3 = qp4[b1 + 1];
      qa[c][0] = f2bf(f0.x * scale); qa[c][1] = f2bf(f0.y * scale);
      qa[c][2] = f2bf(f0.z * scale); qa[c][3] = f2bf(f0.w * scale);
      qa[c][4] = f2bf(f1.x * scale); qa[c][5] = f2bf(f1.y * scale);
      qa[c][6] = f2bf(f1.z * scale); qa[c][7] = f2bf(f1.w * scale);
      qa[c][8]  = f2bf(f2.x * scale); qa[c][9]  = f2bf(f2.y * scale);
      qa[c][10] = f2bf(f2.z * scale); qa[c][11] = f2bf(f2.w * scale);
      qa[c][12] = f2bf(f3.x * scale); qa[c][13] = f2bf(f3.y * scale);
      qa[c][14] = f2bf(f3.z * scale); qa[c][15] = f2bf(f3.w * scale);
    }
  }

  v8f zero = {};
  v8f o[8];
#pragma unroll
  for (int n = 0; n < 8; ++n) o[n] = zero;
  float mrow[8], lrow[8];
#pragma unroll
  for (int f = 0; f < 8; ++f) { mrow[f] = -3.0e38f; lrow[f] = 0.f; }

  const int pmaxb = sp0 + qt * 64 + 63;     // last causal key for this block
  const int nch   = (pmaxb + 32) >> 5;      // 32-key chunks
  const float4 z4 = make_float4(0.f, 0.f, 0.f, 0.f);

  for (int c = 0; c < nch; ++c) {
    const int tb = c << 5;

    // ---- stage K (row-major) and V (transposed) chunks: 32 keys x 128 ----
    // 32 rows x 32 float4 = 1024 float4 each, over 128 threads
#pragma unroll
    for (int j = 0; j < 8; ++j) {
      int idx = j * 128 + tid;
      int r = idx >> 5, c4 = idx & 31;
      int t = tb + r;
      float4 kv = z4, vv = z4;
      if (t < sp0) {
        size_t base = (((size_t)(b * MS + t)) * KVH + kvh) * HD;
        kv = reinterpret_cast<const float4*>(ck + base)[c4];
        vv = reinterpret_cast<const float4*>(cv + base)[c4];
      } else if (t - sp0 < S) {
        size_t base = (((size_t)(b * S + (t - sp0))) * KVH + kvh) * HD;
        kv = reinterpret_cast<const float4*>(kr + base)[c4];
        vv = reinterpret_cast<const float4*>(vr + base)[c4];
      }
      store4bf(&Ks[r][c4 * 4], kv);
      int d0 = c4 * 4;
      Vt[d0 + 0][r] = f2bf(vv.x);
      Vt[d0 + 1][r] = f2bf(vv.y);
      Vt[d0 + 2][r] = f2bf(vv.z);
      Vt[d0 + 3][r] = f2bf(vv.w);
    }
    __syncthreads();

    // ---- scores: two 16x16 tiles, K-contraction over 128 (4 WMMA each) ----
    v8f sc[2];
#pragma unroll
    for (int jt = 0; jt < 2; ++jt) {
      v8f acc = zero;
#pragma unroll
      for (int kc = 0; kc < 4; ++kc) {
        v16bf kb;
#pragma unroll
        for (int i = 0; i < 16; ++i)
          kb[i] = Ks[jt * 16 + ln][kc * 32 + kmapB(i, lane)];
        acc = __builtin_amdgcn_wmma_f32_16x16x32_bf16(
            false, qa[kc], false, kb, (short)0, acc, false, false);
      }
      sc[jt] = acc;
    }

    // ---- causal mask + online softmax (row stats in 16-lane halves) ----
    float al[8];
#pragma unroll
    for (int f = 0; f < 8; ++f) {
      int prow = sp0 + qrow0 + f + 8 * hi;
      float s0 = ((tb + ln)      <= prow) ? sc[0][f] : -1.0e9f;
      float s1 = ((tb + 16 + ln) <= prow) ? sc[1][f] : -1.0e9f;
      float mx = fmaxf(s0, s1);
#pragma unroll
      for (int off = 8; off >= 1; off >>= 1) mx = fmaxf(mx, __shfl_xor(mx, off, 32));
      float mnew = fmaxf(mrow[f], mx);
      float a  = __expf(mrow[f] - mnew);
      float p0 = __expf(s0 - mnew);
      float p1 = __expf(s1 - mnew);
      float rs = p0 + p1;
#pragma unroll
      for (int off = 8; off >= 1; off >>= 1) rs += __shfl_xor(rs, off, 32);
      lrow[f] = lrow[f] * a + rs;
      mrow[f] = mnew;
      al[f]   = a;
      int m = f + 8 * hi;
      Pl[wv_][m][ln]      = f2bf(p0);     // transpose P through LDS
      Pl[wv_][m][16 + ln] = f2bf(p1);
    }

    // rescale O accumulators
#pragma unroll
    for (int n = 0; n < 8; ++n)
#pragma unroll
      for (int f = 0; f < 8; ++f) o[n][f] *= al[f];

    // P as A-fragment (16x32)
    v16bf pa;
#pragma unroll
    for (int i = 0; i < 16; ++i) pa[i] = Pl[wv_][ln][kmapA(i, lane)];

    // ---- P (16x32) x V (32x128): 8 WMMAs across hd columns ----
    // B-fragment from transposed V: two runs of 8 bf16 per lane (k = 8h..8h+7, 16+8h..23+8h)
#pragma unroll
    for (int n = 0; n < 8; ++n) {
      v16bf vb;
      int dcol = n * 16 + ln;
#pragma unroll
      for (int i = 0; i < 16; ++i)
        vb[i] = Vt[dcol][kmapB(i, lane)];
      o[n] = __builtin_amdgcn_wmma_f32_16x16x32_bf16(
          false, pa, false, vb, (short)0, o[n], false, false);
    }
    __syncthreads();   // protect Ks/Vt before next chunk restage
  }

  // ---- normalize and write (B,S,h*128+d) ----
#pragma unroll
  for (int n = 0; n < 8; ++n)
#pragma unroll
    for (int f = 0; f < 8; ++f) {
      int srow = qrow0 + f + 8 * hi;
      out[(((size_t)(b * S + srow)) * H + h) * HD + n * 16 + ln] = o[n][f] / lrow[f];
    }
}

// ---------------------------------------------------------------------------
// Host launcher
// ---------------------------------------------------------------------------
extern "C" void kernel_launch(void* const* d_in, const int* in_sizes, int n_in,
                              void* d_out, int out_size, void* d_ws, size_t ws_size,
                              hipStream_t stream) {
  const float* x  = (const float*)d_in[0];
  const float* wq = (const float*)d_in[1];
  const float* wk = (const float*)d_in[2];
  const float* wv = (const float*)d_in[3];
  const float* wo = (const float*)d_in[4];
  const float* fc = (const float*)d_in[5];
  const float* fs = (const float*)d_in[6];
  const float* ck = (const float*)d_in[7];
  const float* cv = (const float*)d_in[8];
  const int*   sp = (const int*)d_in[9];

  const int D = 4096, H = 32, KVH = 8, HD = 128;
  const int MAXSEQ = in_sizes[5] / (HD / 2);           // freqs_cos: (MAX_SEQ, 64)
  const int B = in_sizes[7] / (MAXSEQ * KVH * HD);     // cache_k: (B,MS,8,128)
  const int S = in_sizes[0] / (B * D);                 // x: (B,S,4096)
  const int M = B * S;

  float* ws   = (float*)d_ws;
  float* q_ws = ws;                                    // M*4096
  float* k_ws = q_ws + (size_t)M * D;                  // M*1024
  float* v_ws = k_ws + (size_t)M * KVH * HD;           // M*1024
  float* a_ws = v_ws + (size_t)M * KVH * HD;           // M*4096

  dim3 gblk(256);
  // QKV projections
  gemm_bf16<<<dim3(D / GT_N, M / GT_M), gblk, 0, stream>>>(x, wq, q_ws, M, D, D);
  gemm_bf16<<<dim3((KVH * HD) / GT_N, M / GT_M), gblk, 0, stream>>>(x, wk, k_ws, M, KVH * HD, D);
  gemm_bf16<<<dim3((KVH * HD) / GT_N, M / GT_M), gblk, 0, stream>>>(x, wv, v_ws, M, KVH * HD, D);

  // RoPE (in place on workspace copies)
  int pq = M * H * (HD / 2);
  rope_inplace<<<(pq + 255) / 256, 256, 0, stream>>>(q_ws, fc, fs, sp, pq, S, H, MAXSEQ);
  int pk = M * KVH * (HD / 2);
  rope_inplace<<<(pk + 255) / 256, 256, 0, stream>>>(k_ws, fc, fs, sp, pk, S, KVH, MAXSEQ);

  // Flash attention: 4 waves / 64 query rows per block
  attn_kernel<<<B * H * (S / 64), 128, 0, stream>>>(q_ws, k_ws, v_ws, ck, cv, sp,
                                                    a_ws, B, S, MAXSEQ);

  // Output projection -> d_out
  gemm_bf16<<<dim3(D / GT_N, M / GT_M), gblk, 0, stream>>>(a_ws, wo, (float*)d_out, M, D, D);
}
